// Qwen3AttentionFused_77816217469138
// MI455X (gfx1250) — matrix operand order repaired and checked by reading, not compiled, not run on hardware
//
#include <hip/hip_runtime.h>
#include <hip/hip_bf16.h>

// ---------------- problem constants ----------------
#define T_   2048
#define HS_  4096
#define H_   32
#define KV_  8
#define D_   128
#define NQKV ((H_ + 2 * KV_) * D_)   // 6144
#define EPS_ 1e-6f
#define ROPE_BASE_ 10000.0f
#define SCALE_ 0.08838834764831845f  // 128^-0.5
#define NEGINF_ (-3.0e38f)

typedef _Float16 half_t;
typedef __attribute__((ext_vector_type(16))) _Float16 v16h;
typedef __attribute__((ext_vector_type(8)))  _Float16 v8h;
typedef __attribute__((ext_vector_type(8)))  float    v8f;
typedef __attribute__((ext_vector_type(4)))  int      v4i_t;

// gfx1250 async global->LDS path (guarded: falls back to VGPR staging)
#if defined(__has_builtin)
#if __has_builtin(__builtin_amdgcn_global_load_async_to_lds_b128) && \
    __has_builtin(__builtin_amdgcn_s_wait_asynccnt)
#define HAS_ASYNC_LDS 1
#endif
#endif
#ifndef HAS_ASYNC_LDS
#define HAS_ASYNC_LDS 0
#endif

#if HAS_ASYNC_LDS
__device__ __forceinline__ void async_cp16(const half_t* g, half_t* l) {
  __builtin_amdgcn_global_load_async_to_lds_b128(
      (__attribute__((address_space(1))) v4i_t*)g,
      (__attribute__((address_space(3))) v4i_t*)l, 0, 0);
}
#endif

// ---------------- WMMA fragment helpers (gfx1250 wave32 layouts, ISA 7.12.2) ----

__device__ __forceinline__ v16h cat8(v8h lo, v8h hi) {
  return __builtin_shufflevector(lo, hi, 0,1,2,3,4,5,6,7,8,9,10,11,12,13,14,15);
}

// A matrix 16x32 f16: lane<16 -> row=lane, elems 0..7 = K0..7, 8..15 = K16..23
//                     lane>=16 -> row=lane-16, elems 0..7 = K8..15, 8..15 = K24..31
__device__ __forceinline__ v16h load_a_frag(const half_t* base, int ld) {
  int lane = threadIdx.x & 31;
  const half_t* p = base + (lane & 15) * ld + ((lane >> 4) << 3);
  v8h lo = *(const v8h*)p;
  v8h hi = *(const v8h*)(p + 16);
  return cat8(lo, hi);
}

// B matrix 32x16 f16, B[k][n] stored as row-major W[n][k]:
// lane<16 -> col=lane, elems 0..7 = K0..7, 8..15 = K16..23 ; lane>=16 -> col=lane-16, elems 0..7 = K8..15, 8..15 = K24..31
__device__ __forceinline__ v16h load_b_frag(const half_t* base, int ld) {
  int lane = threadIdx.x & 31;
  const half_t* p = base + (lane & 15) * ld + ((lane >> 4) << 3);
  v8h lo = *(const v8h*)p;
  v8h hi = *(const v8h*)(p + 16);
  return cat8(lo, hi);
}

__device__ __forceinline__ v8f wmma_f16(v16h a, v16h b, v8f c) {
  return __builtin_amdgcn_wmma_f32_16x16x32_f16(false, a, false, b, (short)0, c,
                                                false, false);
}

// C/D 16x16 f32: lane<16 -> n=lane, vgpr r -> m=r ; lane>=16 -> n=lane-16, m=8+r

// ---------------- kernel 1: RMSNorm(hidden) -> f16 ----------------
__global__ void k_rmsnorm_x(const float* __restrict__ x, const float* __restrict__ w,
                            half_t* __restrict__ xh) {
  int t = blockIdx.x;
  const float* row = x + (size_t)t * HS_;
  __shared__ float red[8];
  float sq = 0.f;
  for (int i = threadIdx.x; i < HS_; i += 256) { float v = row[i]; sq += v * v; }
  #pragma unroll
  for (int off = 1; off < 32; off <<= 1) sq += __shfl_xor(sq, off, 32);
  if ((threadIdx.x & 31) == 0) red[threadIdx.x >> 5] = sq;
  __syncthreads();
  float tot = 0.f;
  #pragma unroll
  for (int i = 0; i < 8; ++i) tot += red[i];
  float r = rsqrtf(tot / (float)HS_ + EPS_);
  for (int i = threadIdx.x; i < HS_; i += 256)
    xh[(size_t)t * HS_ + i] = (half_t)(row[i] * r * w[i]);
}

// ---------------- kernel 2: fp32 -> f16 convert ----------------
__global__ void k_cvt_f16(const float* __restrict__ s, half_t* __restrict__ d, int n) {
  for (int i = blockIdx.x * blockDim.x + threadIdx.x; i < n; i += gridDim.x * blockDim.x)
    d[i] = (half_t)s[i];
}

// ---------------- kernel 3: QKV GEMM + bias + split -> q/k/v f16 ----------------
// out[t,n] = sum_k xh[t,k] * Wqkv[n,k] + bias[n]
// block: 8 waves; wave tile 32x64 (2 A-frags x 4 B-frags -> 8 WMMA / k-step)
__global__ __launch_bounds__(256) void k_qkv_gemm(
    const half_t* __restrict__ X, const half_t* __restrict__ W,
    const float* __restrict__ bias,
    half_t* __restrict__ qb, half_t* __restrict__ kb, half_t* __restrict__ vb) {
  int w  = threadIdx.x >> 5;
  int m0 = blockIdx.y * 64 + (w & 1) * 32;
  int n0 = blockIdx.x * 256 + (w >> 1) * 64;
  v8f c[2][4] = {};
  const half_t* A = X + (size_t)m0 * HS_;
  for (int k = 0; k < HS_; k += 32) {
    __builtin_prefetch(A + k + 256, 0, 1);
    v16h a0 = load_a_frag(A + k, HS_);
    v16h a1 = load_a_frag(A + (size_t)16 * HS_ + k, HS_);
    #pragma unroll
    for (int f = 0; f < 4; ++f) {
      const half_t* B = W + (size_t)(n0 + f * 16) * HS_ + k;
      __builtin_prefetch(B + 256, 0, 1);
      v16h b = load_b_frag(B, HS_);
      c[0][f] = wmma_f16(a0, b, c[0][f]);
      c[1][f] = wmma_f16(a1, b, c[1][f]);
    }
  }
  int lane = threadIdx.x & 31;
  int nl = lane & 15, mb = (lane >> 4) << 3;
  #pragma unroll
  for (int u = 0; u < 2; ++u) {
    #pragma unroll
    for (int f = 0; f < 4; ++f) {
      int n = n0 + f * 16 + nl;
      float b = bias[n];
      #pragma unroll
      for (int r = 0; r < 8; ++r) {
        int t = m0 + u * 16 + mb + r;
        half_t h = (half_t)(c[u][f][r] + b);
        if (n < H_ * D_)                 qb[(size_t)t * (H_ * D_) + n] = h;
        else if (n < (H_ + KV_) * D_)    kb[(size_t)t * (KV_ * D_) + (n - H_ * D_)] = h;
        else                             vb[(size_t)t * (KV_ * D_) + (n - (H_ + KV_) * D_)] = h;
      }
    }
  }
}

// ---------------- kernel 4: per-head RMSNorm + RoPE on q/k (in place, f16) ----
__global__ void k_qk_norm_rope(half_t* __restrict__ qb, half_t* __restrict__ kb,
                               const int* __restrict__ pos,
                               const float* __restrict__ qw,
                               const float* __restrict__ kw) {
  int t = blockIdx.x;
  int hh = blockIdx.y;                  // 0..H+KV-1
  half_t* base; const float* wgt;
  if (hh < H_) { base = qb + ((size_t)t * H_ + hh) * D_; wgt = qw; }
  else         { base = kb + ((size_t)t * KV_ + (hh - H_)) * D_; wgt = kw; }
  __shared__ float xs[D_];
  __shared__ float red[4];
  int tid = threadIdx.x;                // 128 threads
  float v = (float)base[tid];
  float sq = v * v;
  #pragma unroll
  for (int off = 1; off < 32; off <<= 1) sq += __shfl_xor(sq, off, 32);
  if ((tid & 31) == 0) red[tid >> 5] = sq;
  __syncthreads();
  float tot = red[0] + red[1] + red[2] + red[3];
  float r = rsqrtf(tot / (float)D_ + EPS_);
  xs[tid] = v * r * wgt[tid];
  __syncthreads();
  int hd = tid & 63;
  float ang = (float)pos[t] * __expf(-9.210340371976184f * (float)hd * (1.0f / 64.0f));
  float s, c;
  __sincosf(ang, &s, &c);
  float x1 = xs[hd], x2 = xs[hd + 64];
  base[tid] = (half_t)((tid < 64) ? (x1 * c - x2 * s) : (x2 * c + x1 * s));
}

// ---------------- kernel 5: causal GQA flash attention ----------------
// grid: (H, T/128); block: 256 = 8 waves, each wave owns 16 query rows.
__global__ __launch_bounds__(256) void k_attn(
    const half_t* __restrict__ qb, const half_t* __restrict__ kb,
    const half_t* __restrict__ vb, half_t* __restrict__ ob) {
  __shared__ __attribute__((aligned(32))) half_t k_tile[32][D_];  // [key][d]
  __shared__ __attribute__((aligned(32))) half_t vt[D_][32];      // [d][key]
  __shared__ __attribute__((aligned(32))) half_t p_st[8][16][32]; // per-wave P
#if HAS_ASYNC_LDS
  __shared__ __attribute__((aligned(32))) half_t v_row[32][D_];   // [key][d]
#endif

  int h   = blockIdx.x;
  int q0  = blockIdx.y * 128;
  int kvh = h >> 2;                     // H/KV = 4
  int w   = threadIdx.x >> 5;
  int lane = threadIdx.x & 31;
  int qrow = q0 + w * 16;

  // Q fragments: 4 k-chunks of 32 over D=128
  v16h qa[4];
  #pragma unroll
  for (int d = 0; d < 4; ++d)
    qa[d] = load_a_frag(qb + (size_t)qrow * (H_ * D_) + h * D_ + d * 32, H_ * D_);

  float mrow[8], lrow[8];
  #pragma unroll
  for (int r = 0; r < 8; ++r) { mrow[r] = NEGINF_; lrow[r] = 0.f; }
  v8f o[8] = {};

  int nl = lane & 15, mb = (lane >> 4) << 3;
  int jmax = (q0 + 127) >> 5;

  for (int j = 0; j <= jmax; ++j) {
    __syncthreads();
#if HAS_ASYNC_LDS
    {   // async global->LDS copy of K tile and (row-major) V tile
      #pragma unroll
      for (int it = 0; it < 2; ++it) {
        int cchunk = threadIdx.x + it * 256;          // 0..511
        int kr = cchunk >> 4;                         // 0..31
        int cw = (cchunk & 15) * 8;                   // half offset within row
        size_t gro = ((size_t)(j * 32 + kr) * KV_ + kvh) * D_ + cw;
        async_cp16(kb + gro, &k_tile[kr][cw]);
        async_cp16(vb + gro, &v_row[kr][cw]);
      }
      __builtin_amdgcn_s_wait_asynccnt(0);
      __syncthreads();
      // transpose V inside LDS: v_row[k][d] -> vt[d][k]
      int kr = threadIdx.x >> 3;
      int dc = (threadIdx.x & 7) * 16;
      #pragma unroll
      for (int e = 0; e < 16; ++e) vt[dc + e][kr] = v_row[kr][dc + e];
    }
#else
    {   // stage K tile and transposed V tile through VGPRs
      int kr = threadIdx.x >> 3;          // 0..31
      int dc = (threadIdx.x & 7) * 16;    // 0..112
      const half_t* ks = kb + ((size_t)(j * 32 + kr) * KV_ + kvh) * D_ + dc;
      const half_t* vs = vb + ((size_t)(j * 32 + kr) * KV_ + kvh) * D_ + dc;
      #pragma unroll
      for (int e = 0; e < 16; ++e) {
        k_tile[kr][dc + e] = ks[e];
        vt[dc + e][kr]     = vs[e];
      }
    }
#endif
    __syncthreads();

    // S = Q * K^T  (16x32 per wave, two 16x16 C-frags)
    v8f s0 = {}, s1 = {};
    #pragma unroll
    for (int d = 0; d < 4; ++d) {
      s0 = wmma_f16(qa[d], load_b_frag(&k_tile[0][d * 32],  D_), s0);
      s1 = wmma_f16(qa[d], load_b_frag(&k_tile[16][d * 32], D_), s1);
    }

    // online softmax (rows live in 16-lane halves)
    #pragma unroll
    for (int r = 0; r < 8; ++r) {
      int m  = mb + r;
      int qg = qrow + m;
      int kg = j * 32 + nl;
      float a0 = (kg      <= qg) ? s0[r] * SCALE_ : NEGINF_;
      float a1 = (kg + 16 <= qg) ? s1[r] * SCALE_ : NEGINF_;
      float mx = fmaxf(a0, a1);
      #pragma unroll
      for (int off = 1; off < 16; off <<= 1) mx = fmaxf(mx, __shfl_xor(mx, off, 16));
      float mnew = fmaxf(mrow[r], mx);
      float corr = __expf(mrow[r] - mnew);
      mrow[r] = mnew;
      float p0 = __expf(a0 - mnew);
      float p1 = __expf(a1 - mnew);
      float rs = p0 + p1;
      #pragma unroll
      for (int off = 1; off < 16; off <<= 1) rs += __shfl_xor(rs, off, 16);
      lrow[r] = lrow[r] * corr + rs;
      #pragma unroll
      for (int d = 0; d < 8; ++d) o[d][r] *= corr;
      p_st[w][m][nl]      = (half_t)p0;
      p_st[w][m][nl + 16] = (half_t)p1;
    }

    // O += P * V   (A from per-wave LDS stage; B from transposed V tile)
    v16h pa = load_a_frag(&p_st[w][0][0], 32);
    #pragma unroll
    for (int d = 0; d < 8; ++d)
      o[d] = wmma_f16(pa, load_b_frag(&vt[d * 16][0], 32), o[d]);
  }

  // normalize and write out [t, h, d]
  #pragma unroll
  for (int d = 0; d < 8; ++d) {
    #pragma unroll
    for (int r = 0; r < 8; ++r) {
      int m = mb + r;
      float val = o[d][r] / lrow[r];
      ob[(size_t)(qrow + m) * (H_ * D_) + h * D_ + d * 16 + nl] = (half_t)val;
    }
  }
}

// ---------------- kernel 6: O projection GEMM -> fp32 output ----------------
// out[t,n] = sum_j attn[t,j] * Wo[n,j]; wave tile 32x64
__global__ __launch_bounds__(256) void k_o_gemm(
    const half_t* __restrict__ A_, const half_t* __restrict__ W,
    float* __restrict__ out) {
  int w  = threadIdx.x >> 5;
  int m0 = blockIdx.y * 64 + (w & 1) * 32;
  int n0 = blockIdx.x * 256 + (w >> 1) * 64;
  v8f c[2][4] = {};
  const half_t* A = A_ + (size_t)m0 * (H_ * D_);
  for (int k = 0; k < H_ * D_; k += 32) {
    __builtin_prefetch(A + k + 256, 0, 1);
    v16h a0 = load_a_frag(A + k, H_ * D_);
    v16h a1 = load_a_frag(A + (size_t)16 * (H_ * D_) + k, H_ * D_);
    #pragma unroll
    for (int f = 0; f < 4; ++f) {
      const half_t* B = W + (size_t)(n0 + f * 16) * (H_ * D_) + k;
      __builtin_prefetch(B + 256, 0, 1);
      v16h b = load_b_frag(B, H_ * D_);
      c[0][f] = wmma_f16(a0, b, c[0][f]);
      c[1][f] = wmma_f16(a1, b, c[1][f]);
    }
  }
  int lane = threadIdx.x & 31;
  int nl = lane & 15, mb = (lane >> 4) << 3;
  #pragma unroll
  for (int u = 0; u < 2; ++u) {
    #pragma unroll
    for (int f = 0; f < 4; ++f) {
      int n = n0 + f * 16 + nl;
      #pragma unroll
      for (int r = 0; r < 8; ++r)
        out[(size_t)(m0 + u * 16 + mb + r) * HS_ + n] = c[u][f][r];
    }
  }
}

// ---------------- launcher ----------------
extern "C" void kernel_launch(void* const* d_in, const int* in_sizes, int n_in,
                              void* d_out, int out_size, void* d_ws, size_t ws_size,
                              hipStream_t stream) {
  const int*   positions = (const int*)  d_in[0];
  const float* hidden    = (const float*)d_in[1];
  const float* ln_w      = (const float*)d_in[2];
  const float* qkv_w     = (const float*)d_in[3];
  const float* qkv_b     = (const float*)d_in[4];
  const float* qn_w      = (const float*)d_in[5];
  const float* kn_w      = (const float*)d_in[6];
  const float* o_w       = (const float*)d_in[7];
  float* out = (float*)d_out;

  // workspace carve-up (f16 elements)
  half_t* ws    = (half_t*)d_ws;
  half_t* xh    = ws;                              // T*HS
  half_t* wqkvh = xh    + (size_t)T_ * HS_;        // NQKV*HS
  half_t* woh   = wqkvh + (size_t)NQKV * HS_;      // HS*HS
  half_t* qb    = woh   + (size_t)HS_ * HS_;       // T*H*D
  half_t* kb    = qb    + (size_t)T_ * H_ * D_;    // T*KV*D
  half_t* vb    = kb    + (size_t)T_ * KV_ * D_;   // T*KV*D
  half_t* ab    = vb    + (size_t)T_ * KV_ * D_;   // T*H*D

  // 1. weight conversions + input norm
  k_cvt_f16<<<4096, 256, 0, stream>>>(qkv_w, wqkvh, NQKV * HS_);
  k_cvt_f16<<<4096, 256, 0, stream>>>(o_w, woh, HS_ * HS_);
  k_rmsnorm_x<<<T_, 256, 0, stream>>>(hidden, ln_w, xh);

  // 2. QKV projection (block tile 64x256)
  k_qkv_gemm<<<dim3(NQKV / 256, T_ / 64), 256, 0, stream>>>(xh, wqkvh, qkv_b, qb, kb, vb);

  // 3. q/k RMSNorm + RoPE
  k_qk_norm_rope<<<dim3(T_, H_ + KV_), 128, 0, stream>>>(qb, kb, positions, qn_w, kn_w);

  // 4. attention
  k_attn<<<dim3(H_, T_ / 128), 256, 0, stream>>>(qb, kb, vb, ab);

  // 5. output projection (block tile 64x256)
  k_o_gemm<<<dim3(HS_ / 256, T_ / 64), 256, 0, stream>>>(ab, woh, out);
}
